// DSIGF_43155831390419
// MI455X (gfx1250) — matrix operand order repaired and checked by reading, not compiled, not run on hardware
//
#include <hip/hip_runtime.h>
#include <math.h>

#define BB     8
#define DD     512
#define L0C    1024
#define LL     1025
#define LPAD   1040            // 65 * 16
#define MROWS  (BB * LPAD)     // 8320 rows, multiple of 128
#define HH     8
#define DHH    64
#define DFFC   2048
#define EPSC   1e-6f
#define SCALEC 0.125f          // 1/sqrt(64)

typedef __attribute__((ext_vector_type(16))) __bf16 v16bf;
typedef __attribute__((ext_vector_type(8)))  float  v8f;
typedef unsigned int u32x4 __attribute__((ext_vector_type(4)));
typedef int          i32x4 __attribute__((ext_vector_type(4)));
typedef int          i32x8 __attribute__((ext_vector_type(8)));

#if defined(__has_builtin)
# if __has_builtin(__builtin_amdgcn_tensor_load_to_lds) && \
     __has_builtin(__builtin_amdgcn_s_wait_tensorcnt)
#  define HAVE_TDM 1
# endif
#endif
#ifndef HAVE_TDM
# define HAVE_TDM 0
#endif

#if HAVE_TDM
# if defined(__clang_major__) && (__clang_major__ >= 23)
#  pragma message("CDNA5 probe: TDM tensor_load_to_lds ENABLED (6-arg, clang>=23)")
# else
#  pragma message("CDNA5 probe: TDM tensor_load_to_lds ENABLED (5-arg, clang<23)")
# endif
#else
# pragma message("CDNA5 probe: TDM builtins NOT found - using vector-load fallback")
#endif

union BfPair { unsigned u; __bf16 h[2]; };

static __device__ __forceinline__ unsigned short f2bf_bits(float f) {
  __bf16 h = (__bf16)f;
  return __builtin_bit_cast(unsigned short, h);
}

#if HAVE_TDM
// Issue a TDM DMA of a 128-row x 32-col bf16 tile of A (row stride K elements)
// into LDS at byte offset ldsa. Whole-wave instruction; EXEC is ignored.
static __device__ __forceinline__ void
tdm_load_a_tile(const __bf16* A, long m_base, int K, int kb, unsigned ldsa)
{
  const unsigned long long ga =
      (unsigned long long)(const void*)(A + (long)m_base * K + kb);
  u32x4 g0;
  g0[0] = 1u;                                          // count=1, user mode
  g0[1] = ldsa;                                        // lds_addr (bytes)
  g0[2] = (unsigned)(ga & 0xffffffffu);                // global_addr[31:0]
  g0[3] = (unsigned)((ga >> 32) & 0x01ffffffu)         // global_addr[56:32]
          | (2u << 30);                                // type = 2 (image)
  i32x8 g1;
  g1[0] = (int)(1u << 16);                             // data_size = 2 bytes
  g1[1] = (int)(((unsigned)K & 0xffffu) << 16);        // tensor_dim0 lo16
  g1[2] = (int)(((unsigned)K >> 16) |
                (((unsigned)MROWS & 0xffffu) << 16));  // dim0 hi / dim1 lo
  g1[3] = (int)(((unsigned)MROWS >> 16) | (32u << 16));// dim1 hi, tile_dim0=32
  g1[4] = (int)(128u);                                 // tile_dim1=128, dim2=0
  g1[5] = (int)(unsigned)K;                            // dim0_stride lo32
  g1[6] = 0;                                           // stride hi, dim1_stride
  g1[7] = 0;
  i32x4 z4 = {0, 0, 0, 0};
#if defined(__clang_major__) && (__clang_major__ >= 23)
  i32x8 z8 = {0, 0, 0, 0, 0, 0, 0, 0};
  __builtin_amdgcn_tensor_load_to_lds(g0, g1, z4, z4, z8, 0);
#else
  __builtin_amdgcn_tensor_load_to_lds(g0, g1, z4, z4, 0);
#endif
}
#endif

// ---------------------------------------------------------------------------
// Build x1 (f32 + bf16) and x2 (bf16): transpose embed [B,D,L0] -> [B,L,D],
// prepend cls at l==0, zero the padded rows l in [1025,1040).
// ---------------------------------------------------------------------------
__global__ void __launch_bounds__(128)
prep_kernel(const float* __restrict__ e1, const float* __restrict__ e2,
            const float* __restrict__ cls, float* __restrict__ xf,
            __bf16* __restrict__ xb, __bf16* __restrict__ x2b)
{
  const int row = blockIdx.x;            // 0 .. MROWS-1
  const int b = row / LPAD, l = row % LPAD;
  const int tid = threadIdx.x;
#pragma unroll
  for (int i = 0; i < 4; ++i) {
    const int d = tid * 4 + i;
    float v1, v2;
    if (l == 0)            { v1 = cls[d]; v2 = v1; }
    else if (l <= L0C)     { const long o = ((long)b * DD + d) * L0C + (l - 1);
                             v1 = e1[o]; v2 = e2[o]; }
    else                   { v1 = 0.0f; v2 = 0.0f; }
    const long o = (long)row * DD + d;
    xf[o]  = v1;
    xb[o]  = (__bf16)v1;
    x2b[o] = (__bf16)v2;
  }
}

// ---------------------------------------------------------------------------
// Tiled WMMA GEMM:  out[M,N] = A[M,K](bf16) * W[N,K](f32->bf16)^T + bias[N]
// Block = 128 threads (4 waves); tile 128x64; each wave owns 32x64 (8 WMMA
// accumulators, B-fragments reused across the two 16-row strips).
// A-tiles are double-buffered in LDS and staged by the Tensor Data Mover:
// the DMA for tile k+1 is issued before computing on tile k, and the wave
// only waits for TENSORcnt<=1 (TDM ops complete in order). W-tile is
// converted f32->bf16 by VALU.
// EPI: 0 -> store f32, 1 -> store bf16, 2 -> exact GELU then store bf16.
// ---------------------------------------------------------------------------
template<int EPI>
__global__ void __launch_bounds__(128)
gemm_wmma(const __bf16* __restrict__ A, const float* __restrict__ W,
          const float* __restrict__ bias, float* __restrict__ outF,
          __bf16* __restrict__ outB, int K)
{
  __shared__ __align__(16) unsigned short a_sh[2][128 * 32];
  __shared__ __align__(16) unsigned short w_sh[64 * 32];

  const int tid    = threadIdx.x;
  const int lane   = tid & 31;
  const int wv     = tid >> 5;       // wave id 0..3 -> 32-row strip
  const int hf     = lane >> 4;      // half-wave id
  const int ln     = lane & 15;
  const int m_base = blockIdx.x * 128;
  const int n_base = blockIdx.y * 64;
  const int N      = (int)gridDim.y * 64;

  v8f acc[2][4] = { { {}, {}, {}, {} }, { {}, {}, {}, {} } };

#if HAVE_TDM
  const unsigned ldsa0 = (unsigned)(__SIZE_TYPE__)(const void*)&a_sh[0][0];
  const unsigned ldsa1 = (unsigned)(__SIZE_TYPE__)(const void*)&a_sh[1][0];
  if (tid < 32) tdm_load_a_tile(A, m_base, K, 0, ldsa0);   // prime buffer 0
#endif

  for (int kb = 0; kb < K; kb += 32) {
    const int cur = (kb >> 5) & 1;

#if HAVE_TDM
    // issue DMA for the next tile into the alternate buffer (async overlap)
    if (tid < 32 && kb + 32 < K)
      tdm_load_a_tile(A, m_base, K, kb + 32, cur ? ldsa0 : ldsa1);
#endif

    // ---- stage W tile (64n x 32k), convert f32 -> bf16 ----
    {
      const int wrow = tid >> 1, whalf = tid & 1;
      const float* wpt = W + (long)(n_base + wrow) * K + kb + whalf * 16;
      if (kb + 32 < K) __builtin_prefetch(wpt + 32, 0, 3);
#pragma unroll
      for (int i = 0; i < 16; i += 2) {
        BfPair p; p.h[0] = (__bf16)wpt[i]; p.h[1] = (__bf16)wpt[i + 1];
        *(unsigned*)&w_sh[wrow * 32 + whalf * 16 + i] = p.u;
      }
    }

#if HAVE_TDM
    // wait until the DMA for the *current* buffer has landed (in-order TDM)
    if (kb + 32 < K) { if (tid < 32) __builtin_amdgcn_s_wait_tensorcnt(1); }
    else             { if (tid < 32) __builtin_amdgcn_s_wait_tensorcnt(0); }
#else
    // ---- fallback: cooperative vector loads (1 row of 32 bf16 per thread) ----
    {
      const __bf16* ap = A + (long)(m_base + tid) * K + kb;
      uint4 v0 = *(const uint4*)(ap);
      uint4 v1 = *(const uint4*)(ap + 8);
      uint4 v2 = *(const uint4*)(ap + 16);
      uint4 v3 = *(const uint4*)(ap + 24);
      *(uint4*)&a_sh[0][tid * 32 + 0]  = v0;
      *(uint4*)&a_sh[0][tid * 32 + 8]  = v1;
      *(uint4*)&a_sh[0][tid * 32 + 16] = v2;
      *(uint4*)&a_sh[0][tid * 32 + 24] = v3;
    }
#endif
    __syncthreads();

#if HAVE_TDM
    const unsigned short* asrc = &a_sh[cur][0];
#else
    const unsigned short* asrc = &a_sh[0][0];
#endif

    // ---- A fragments for the wave's two 16-row strips ----
    v16bf af0, af1;
#pragma unroll
    for (int i = 0; i < 8; ++i) {
      const int k0 = hf * 16 + 2 * i;
      BfPair p0; p0.u = *(const unsigned*)&asrc[(wv * 32 + ln) * 32 + k0];
      BfPair p1; p1.u = *(const unsigned*)&asrc[(wv * 32 + 16 + ln) * 32 + k0];
      af0[2 * i] = p0.h[0]; af0[2 * i + 1] = p0.h[1];
      af1[2 * i] = p1.h[0]; af1[2 * i + 1] = p1.h[1];
    }

#pragma unroll
    for (int j = 0; j < 4; ++j) {
      // B fragment: B[k][n] = W[n][k]; lane ln -> column n, half -> k 0-15/16-31
      v16bf bfg;
#pragma unroll
      for (int i = 0; i < 8; ++i) {
        const int c0 = hf * 16 + 2 * i;
        BfPair p; p.u = *(const unsigned*)&w_sh[(j * 16 + ln) * 32 + c0];
        bfg[2 * i] = p.h[0]; bfg[2 * i + 1] = p.h[1];
      }
      acc[0][j] = __builtin_amdgcn_wmma_f32_16x16x32_bf16(
                    false, af0, false, bfg, (short)0, acc[0][j], false, false);
      acc[1][j] = __builtin_amdgcn_wmma_f32_16x16x32_bf16(
                    false, af1, false, bfg, (short)0, acc[1][j], false, false);
    }
    __syncthreads();
  }

  // ---- epilogue: C/D layout row = r + 8*hf, col = ln ----
#pragma unroll
  for (int s = 0; s < 2; ++s) {
#pragma unroll
    for (int j = 0; j < 4; ++j) {
      const int col = n_base + j * 16 + ln;
      const float bc = bias[col];
#pragma unroll
      for (int r = 0; r < 8; ++r) {
        const int row = m_base + wv * 32 + s * 16 + r + hf * 8;
        float v = acc[s][j][r] + bc;
        if (EPI == 2) v = 0.5f * v * (1.0f + erff(v * 0.70710678118654752f));
        const long o = (long)row * N + col;
        if (EPI == 0) outF[o] = v;
        else          outB[o] = (__bf16)v;
      }
    }
  }
}

// ---------------------------------------------------------------------------
// Flash attention: one wave per (b, h, 16-row q tile); streams 32-key tiles.
// scores = Q K^T * SCALE (2 WMMA), online softmax, ctx += P V (4 WMMA).
// Keys >= 1025 are masked to -inf; V for masked keys forced to 0 (NaN guard).
// ---------------------------------------------------------------------------
__global__ void __launch_bounds__(32)
flash_attn(const __bf16* __restrict__ Q, const __bf16* __restrict__ Km,
           const __bf16* __restrict__ Vm, __bf16* __restrict__ Ctx)
{
  __shared__ __align__(4) unsigned short p_sh[16 * 32];
  const int lane = threadIdx.x;
  const int hf = lane >> 4, ln = lane & 15;
  const int qt = blockIdx.x;
  const int h  = blockIdx.y;
  const int b  = blockIdx.z;
  const long rowBase = (long)b * LPAD;
  const int  qr = qt * 16;

  // Q fragments for dims 0..31 and 32..63 (A-matrix layout)
  v16bf qa0, qa1;
  {
    const __bf16* qp = Q + (rowBase + qr + ln) * DD + h * DHH;
#pragma unroll
    for (int i = 0; i < 8; ++i) {
      const int k0 = hf * 16 + 2 * i;
      BfPair p0; p0.u = *(const unsigned*)(qp + k0);
      BfPair p1; p1.u = *(const unsigned*)(qp + 32 + k0);
      qa0[2 * i] = p0.h[0]; qa0[2 * i + 1] = p0.h[1];
      qa1[2 * i] = p1.h[0]; qa1[2 * i + 1] = p1.h[1];
    }
  }

  float mrun[8], lrun[8];
#pragma unroll
  for (int r = 0; r < 8; ++r) { mrun[r] = -__builtin_inff(); lrun[r] = 0.0f; }
  v8f ca[4] = { {}, {}, {}, {} };

  for (int kt = 0; kt < 33; ++kt) {   // 33*32 = 1056 >= 1025 keys
    const int kb0 = kt * 32;
    v8f s[2];
#pragma unroll
    for (int nt = 0; nt < 2; ++nt) {
      const int key = kb0 + nt * 16 + ln;  // this lane's key column
      v16bf bk0 = {}; v16bf bk1 = {};
      if (key < LL) {
        const __bf16* kp = Km + (rowBase + key) * DD + h * DHH;
        if (key + 32 < LL)
          __builtin_prefetch(kp + 32 * DD, 0, 3);     // next key tile, same lane
#pragma unroll
        for (int i = 0; i < 8; ++i) {
          const int c0 = hf * 16 + 2 * i;
          BfPair p0; p0.u = *(const unsigned*)(kp + c0);
          BfPair p1; p1.u = *(const unsigned*)(kp + 32 + c0);
          bk0[2 * i] = p0.h[0]; bk0[2 * i + 1] = p0.h[1];
          bk1[2 * i] = p1.h[0]; bk1[2 * i + 1] = p1.h[1];
        }
      }
      v8f sa = {};
      sa = __builtin_amdgcn_wmma_f32_16x16x32_bf16(false, qa0, false, bk0, (short)0, sa, false, false);
      sa = __builtin_amdgcn_wmma_f32_16x16x32_bf16(false, qa1, false, bk1, (short)0, sa, false, false);
      s[nt] = sa;
    }

    const int col0 = kb0 + ln;
    const int col1 = kb0 + 16 + ln;
#pragma unroll
    for (int r = 0; r < 8; ++r) {
      float v0 = s[0][r] * SCALEC; if (col0 >= LL) v0 = -__builtin_inff();
      float v1 = s[1][r] * SCALEC; if (col1 >= LL) v1 = -__builtin_inff();
      float t = fmaxf(v0, v1);
#pragma unroll
      for (int off = 1; off < 16; off <<= 1) t = fmaxf(t, __shfl_xor(t, off, 32));
      const float mn = fmaxf(mrun[r], t);
      const float al = expf(mrun[r] - mn);
      const float p0 = expf(v0 - mn);
      const float p1 = expf(v1 - mn);
      float rs = p0 + p1;
#pragma unroll
      for (int off = 1; off < 16; off <<= 1) rs += __shfl_xor(rs, off, 32);
      lrun[r] = lrun[r] * al + rs;
      mrun[r] = mn;
#pragma unroll
      for (int j = 0; j < 4; ++j) ca[j][r] *= al;
      const int prow = r + hf * 8;               // C/D layout row
      p_sh[prow * 32 + ln]      = f2bf_bits(p0);
      p_sh[prow * 32 + 16 + ln] = f2bf_bits(p1);
    }
    __syncthreads();

    // P as A-matrix fragment (16x32)
    v16bf pa;
#pragma unroll
    for (int i = 0; i < 8; ++i) {
      const int k0 = hf * 16 + 2 * i;
      BfPair p; p.u = *(const unsigned*)&p_sh[ln * 32 + k0];
      pa[2 * i] = p.h[0]; pa[2 * i + 1] = p.h[1];
    }
    // ctx += P * V  (V fragments per 16-dim subtile)
    {
      const int vkey = kb0 + hf * 16;
      if (vkey + 32 < LL)
        __builtin_prefetch(Vm + (rowBase + vkey + 32) * DD + h * DHH, 0, 3);
    }
#pragma unroll
    for (int j = 0; j < 4; ++j) {
      v16bf bv = {};
#pragma unroll
      for (int e = 0; e < 16; ++e) {
        const int key = kb0 + hf * 16 + e;
        if (key < LL) bv[e] = Vm[(rowBase + key) * DD + h * DHH + j * 16 + ln];
      }
      ca[j] = __builtin_amdgcn_wmma_f32_16x16x32_bf16(false, pa, false, bv, (short)0, ca[j], false, false);
    }
    __syncthreads();
  }

#pragma unroll
  for (int j = 0; j < 4; ++j) {
#pragma unroll
    for (int r = 0; r < 8; ++r) {
      const int row = qr + r + hf * 8;
      const float v = ca[j][r] / lrun[r];
      Ctx[(rowBase + row) * DD + h * DHH + j * 16 + ln] = (__bf16)v;
    }
  }
}

// ---------------------------------------------------------------------------
// x = LayerNorm(x + t) * g + b ; refresh both f32 master and bf16 copy.
// One 128-thread block per row (512 elems, 4 per thread).
// ---------------------------------------------------------------------------
__global__ void __launch_bounds__(128)
ln_residual(float* __restrict__ x, const float* __restrict__ t,
            const float* __restrict__ g, const float* __restrict__ bta,
            __bf16* __restrict__ xb)
{
  __shared__ float red[4];
  const int row = blockIdx.x;
  const int tid = threadIdx.x;
  const int lane = tid & 31, wv = tid >> 5;
  const long base = (long)row * DD + tid * 4;

  float v[4];
  float s = 0.0f;
#pragma unroll
  for (int i = 0; i < 4; ++i) { v[i] = x[base + i] + t[base + i]; s += v[i]; }
#pragma unroll
  for (int off = 1; off < 32; off <<= 1) s += __shfl_xor(s, off, 32);
  if (lane == 0) red[wv] = s;
  __syncthreads();
  const float mean = (red[0] + red[1] + red[2] + red[3]) * (1.0f / DD);
  __syncthreads();

  float s2 = 0.0f;
#pragma unroll
  for (int i = 0; i < 4; ++i) { const float d = v[i] - mean; s2 += d * d; }
#pragma unroll
  for (int off = 1; off < 32; off <<= 1) s2 += __shfl_xor(s2, off, 32);
  if (lane == 0) red[wv] = s2;
  __syncthreads();
  const float var = (red[0] + red[1] + red[2] + red[3]) * (1.0f / DD);
  const float inv = rsqrtf(var + EPSC);

#pragma unroll
  for (int i = 0; i < 4; ++i) {
    const int d = tid * 4 + i;
    const float y = (v[i] - mean) * inv * g[d] + bta[d];
    x[base + i]  = y;
    xb[base + i] = (__bf16)y;
  }
}

// ---------------------------------------------------------------------------
// Copy valid [B,1025,512] region of padded x into d_out.
// ---------------------------------------------------------------------------
__global__ void copy_out(const float* __restrict__ xf, float* __restrict__ out)
{
  const long i = (long)blockIdx.x * blockDim.x + threadIdx.x;
  const long total = (long)BB * LL * DD;
  if (i >= total) return;
  const int d = (int)(i % DD);
  const long rl = i / DD;
  const int l = (int)(rl % LL);
  const int b = (int)(rl / LL);
  out[i] = xf[((long)b * LPAD + l) * DD + d];
}

// ---------------------------------------------------------------------------
extern "C" void kernel_launch(void* const* d_in, const int* in_sizes, int n_in,
                              void* d_out, int out_size, void* d_ws, size_t ws_size,
                              hipStream_t stream)
{
  (void)in_sizes; (void)n_in; (void)out_size; (void)ws_size;
  const float* embed1 = (const float*)d_in[0];
  const float* embed2 = (const float*)d_in[1];
  const float* cls    = (const float*)d_in[2];
  const float* Wq = (const float*)d_in[3];
  const float* bq = (const float*)d_in[4];
  const float* Wk = (const float*)d_in[5];
  const float* bk = (const float*)d_in[6];
  const float* Wv = (const float*)d_in[7];
  const float* bv = (const float*)d_in[8];
  const float* Wo = (const float*)d_in[9];
  const float* bo = (const float*)d_in[10];
  const float* ln1g = (const float*)d_in[11];
  const float* ln1b = (const float*)d_in[12];
  const float* W1 = (const float*)d_in[13];
  const float* b1 = (const float*)d_in[14];
  const float* W2 = (const float*)d_in[15];
  const float* b2 = (const float*)d_in[16];
  const float* ln2g = (const float*)d_in[17];
  const float* ln2b = (const float*)d_in[18];
  float* out = (float*)d_out;

  // workspace carve (~114 MB), 256B aligned slabs
  char* wp = (char*)d_ws;
  auto carve = [&](size_t bytes) -> void* {
    char* p = wp; wp += (bytes + 255) & ~(size_t)255; return (void*)p;
  };
  float*  xf   = (float*) carve((size_t)MROWS * DD * 4);
  __bf16* xb   = (__bf16*)carve((size_t)MROWS * DD * 2);
  __bf16* x2b  = (__bf16*)carve((size_t)MROWS * DD * 2);
  __bf16* qb   = (__bf16*)carve((size_t)MROWS * DD * 2);
  __bf16* kb   = (__bf16*)carve((size_t)MROWS * DD * 2);
  __bf16* vb   = (__bf16*)carve((size_t)MROWS * DD * 2);
  __bf16* ctxb = (__bf16*)carve((size_t)MROWS * DD * 2);
  float*  tmp  = (float*) carve((size_t)MROWS * DD * 4);
  __bf16* hb   = (__bf16*)carve((size_t)MROWS * DFFC * 2);

  prep_kernel<<<dim3(MROWS), dim3(128), 0, stream>>>(embed1, embed2, cls, xf, xb, x2b);

  const dim3 gD (MROWS / 128, DD   / 64);  // 65 x 8
  const dim3 gFF(MROWS / 128, DFFC / 64);  // 65 x 32
  const dim3 blk(128);
  const dim3 gFA(LPAD / 16, HH, BB);       // 65 x 8 x 8 flash tiles

  for (int l = 0; l < 2; ++l) {
    const float* Wq_l = Wq + (size_t)l * DD * DD;
    const float* Wk_l = Wk + (size_t)l * DD * DD;
    const float* Wv_l = Wv + (size_t)l * DD * DD;
    const float* Wo_l = Wo + (size_t)l * DD * DD;

    gemm_wmma<1><<<gD, blk, 0, stream>>>(xb,  Wq_l, bq + l * DD, nullptr, qb, DD);
    gemm_wmma<1><<<gD, blk, 0, stream>>>(x2b, Wk_l, bk + l * DD, nullptr, kb, DD);
    gemm_wmma<1><<<gD, blk, 0, stream>>>(x2b, Wv_l, bv + l * DD, nullptr, vb, DD);

    flash_attn<<<gFA, dim3(32), 0, stream>>>(qb, kb, vb, ctxb);

    gemm_wmma<0><<<gD, blk, 0, stream>>>(ctxb, Wo_l, bo + l * DD, tmp, nullptr, DD);
    ln_residual<<<dim3(MROWS), dim3(128), 0, stream>>>(xf, tmp, ln1g + l * DD, ln1b + l * DD, xb);

    gemm_wmma<2><<<gFF, blk, 0, stream>>>(xb, W1 + (size_t)l * DFFC * DD, b1 + l * DFFC,
                                          nullptr, hb, DD);
    gemm_wmma<0><<<gD, blk, 0, stream>>>(hb, W2 + (size_t)l * DD * DFFC, b2 + l * DD,
                                         tmp, nullptr, DFFC);
    ln_residual<<<dim3(MROWS), dim3(128), 0, stream>>>(xf, tmp, ln2g + l * DD, ln2b + l * DD, xb);
  }

  copy_out<<<dim3((BB * LL * DD + 255) / 256), dim3(256), 0, stream>>>(xf, out);
}
